// LSTMModel_40252433498406
// MI455X (gfx1250) — compile-verified, matrix operand order checked by reading, not run on hardware
//
#include <hip/hip_runtime.h>

// ---------------------------------------------------------------------------
// Fused LSTM (T=1501, H=128, B=512, in=1) + dense(T*H -> 1) + sigmoid.
//
// Strategy (MI455X / gfx1250, wave32):
//  * 32 workgroups x 16 batch rows; recurrence runs entirely inside one WG
//    (no global sync; c-state in VGPRs, h-state in LDS as bf16).
//  * Per step: g(16x512) = h(16x128) @ W_hh^T(128x512) via
//    v_wmma_f32_16x16x32_bf16. 8 waves, 4 N-tiles each; W_hh^T fragments are
//    built once and kept register-resident (16 x v16bf per wave).
//  * Dense layer folded into the time loop (dacc += h_t . dense_w[t,:]) so
//    the 393MB hidden-state tensor is never materialized; total HBM traffic
//    is a few MB (L2-resident), irrelevant at 23.3 TB/s -> latency-bound.
//  * x[b,t] and dense_w[t,:] slices prefetched at step top, consumed after
//    the barrier, hiding global latency under the WMMA phase.
// ---------------------------------------------------------------------------

typedef __attribute__((ext_vector_type(16))) __bf16 v16bf;
typedef __attribute__((ext_vector_type(8)))  float  v8f;

#define T_LEN   1501
#define HID     128
#define BATCH   512
#define BTILE   16     // batch rows per workgroup
#define THREADS 256    // 8 waves

__device__ __forceinline__ float sigm(float x)      { return 1.0f / (1.0f + __expf(-x)); }
__device__ __forceinline__ float tanh_fast(float x) { return 1.0f - 2.0f / (1.0f + __expf(2.0f * x)); }

__device__ __forceinline__ void load8(float* d, const float* s) {
    float4 a = *(const float4*)(s);
    float4 b = *(const float4*)(s + 4);
    d[0]=a.x; d[1]=a.y; d[2]=a.z; d[3]=a.w;
    d[4]=b.x; d[5]=b.y; d[6]=b.z; d[7]=b.w;
}

__global__ __launch_bounds__(THREADS)
void lstm_fused_kernel(const float* __restrict__ x,      // (B,1,T)
                       const float* __restrict__ W_ih,   // (4H,1)
                       const float* __restrict__ W_hh,   // (4H,H)
                       const float* __restrict__ b_ih,   // (4H)
                       const float* __restrict__ b_hh,   // (4H)
                       const float* __restrict__ dw,     // (1, T*H)
                       const float* __restrict__ db,     // (1)
                       float* __restrict__ out)          // (B,1)
{
    // LDS. Strides padded to avoid bank conflicts:
    //  hbuf: 136 ushorts/row -> lane m's b128 read hits banks 4m..4m+3
    //  gbuf: 516 floats/row  -> lane-half row groups land on disjoint banks
    __shared__ __align__(16) unsigned short hbuf[BTILE][136]; // h state, bf16
    __shared__ __align__(16) float          gbuf[BTILE][516]; // gate pre-acts
    __shared__ __align__(16) float          wih_s[4 * HID];
    __shared__ __align__(16) float          bias_s[4 * HID];
    __shared__ float                        red[THREADS];

    const int tid  = threadIdx.x;
    const int lane = tid & 31;
    const int wv   = tid >> 5;        // wave 0..7
    const int lcol = lane & 15;       // column within a 16-wide tile / A row M
    const int half = lane >> 4;       // lane half selects K sub-ranges

    // gate-phase mapping: 16*128 state elems, 8 contiguous h per thread
    const int gb = tid >> 4;              // local batch row 0..15
    const int h0 = (tid & 15) * 8;        // hidden start
    const int bg = blockIdx.x * BTILE + gb;

    // ---- one-time staging: W_ih, combined bias, zero h state ----
    for (int j = tid; j < 4 * HID; j += THREADS) {
        wih_s[j]  = W_ih[j];
        bias_s[j] = b_ih[j] + b_hh[j];
    }
    for (int j = tid; j < BTILE * 136; j += THREADS)
        ((unsigned short*)hbuf)[j] = 0;

    // ---- one-time: register-resident W_hh^T bf16 fragments -------------
    // Wave wv owns output columns [wv*64, wv*64+64): 4 n-tiles x 4 k-chunks.
    // 16-bit B layout (mirror of A): lane holds column n; element e maps to
    // k = kc*32 + 8*half + (e<8 ? e : e+8).
    v16bf Bfrag[4][4];
    #pragma unroll
    for (int nt = 0; nt < 4; ++nt) {
        const int n = wv * 64 + nt * 16 + lcol;   // row of W_hh (gate unit)
        #pragma unroll
        for (int kc = 0; kc < 4; ++kc) {
            const int kbase = kc * 32 + 8 * half;
            v16bf bfr;
            #pragma unroll
            for (int e = 0; e < 16; ++e) {
                const int k = kbase + ((e < 8) ? e : (e + 8));
                bfr[e] = (__bf16)W_hh[n * HID + k];
            }
            Bfrag[nt][kc] = bfr;
        }
    }

    float c_state[8];
    #pragma unroll
    for (int j = 0; j < 8; ++j) c_state[j] = 0.0f;
    float dacc = 0.0f;

    __syncthreads();

    for (int t = 0; t < T_LEN; ++t) {
        // Prefetch this step's scalar input and dense_w slice; the loadcnt
        // wait lands after the WMMA phase, hiding L2 latency.
        const float  xv  = x[bg * T_LEN + t];
        const float4 dwa = *(const float4*)&dw[t * HID + h0];
        const float4 dwb = *(const float4*)&dw[t * HID + h0 + 4];

        // ---- WMMA phase: g = h @ W_hh^T ----
        const v8f zero = {0.f, 0.f, 0.f, 0.f, 0.f, 0.f, 0.f, 0.f};
        v8f acc[4] = {zero, zero, zero, zero};

        #pragma unroll
        for (int kc = 0; kc < 4; ++kc) {
            // A fragment: lane = row M, element e -> k = kc*32+8*half+(e<8?e:e+8)
            union { uint4 u[2]; v16bf v; } af;
            const int kA = kc * 32 + 8 * half;
            af.u[0] = *(const uint4*)&hbuf[lcol][kA];
            af.u[1] = *(const uint4*)&hbuf[lcol][kA + 16];
            #pragma unroll
            for (int nt = 0; nt < 4; ++nt) {
                acc[nt] = __builtin_amdgcn_wmma_f32_16x16x32_bf16(
                    false, af.v, false, Bfrag[nt][kc],
                    (short)0, acc[nt], false, false);
            }
        }

        // C/D layout: VGPR r of lane -> row (r + 8*half), col lcol (in tile)
        #pragma unroll
        for (int nt = 0; nt < 4; ++nt) {
            const int col = wv * 64 + nt * 16 + lcol;
            #pragma unroll
            for (int r = 0; r < 8; ++r)
                gbuf[8 * half + r][col] = acc[nt][r];
        }
        __syncthreads();

        // ---- gate phase: 8 (b,h) elements per thread ----
        float iv[8], fv[8], gv[8], ov[8];
        load8(iv, &gbuf[gb][0 * HID + h0]);
        load8(fv, &gbuf[gb][1 * HID + h0]);
        load8(gv, &gbuf[gb][2 * HID + h0]);
        load8(ov, &gbuf[gb][3 * HID + h0]);

        float wi[8], wf[8], wg[8], wo[8];
        load8(wi, &wih_s[0 * HID + h0]);
        load8(wf, &wih_s[1 * HID + h0]);
        load8(wg, &wih_s[2 * HID + h0]);
        load8(wo, &wih_s[3 * HID + h0]);

        float bi[8], bf[8], bgt[8], bo[8];
        load8(bi,  &bias_s[0 * HID + h0]);
        load8(bf,  &bias_s[1 * HID + h0]);
        load8(bgt, &bias_s[2 * HID + h0]);
        load8(bo,  &bias_s[3 * HID + h0]);

        const float dwv[8] = {dwa.x, dwa.y, dwa.z, dwa.w, dwb.x, dwb.y, dwb.z, dwb.w};

        union { unsigned short s[8]; uint4 u; } hnew;
        #pragma unroll
        for (int j = 0; j < 8; ++j) {
            const float ii = sigm(iv[j] + xv * wi[j] + bi[j]);
            const float ff = sigm(fv[j] + xv * wf[j] + bf[j]);
            const float gg = tanh_fast(gv[j] + xv * wg[j] + bgt[j]);
            const float oo = sigm(ov[j] + xv * wo[j] + bo[j]);
            const float c  = ff * c_state[j] + ii * gg;
            c_state[j] = c;
            const float h = oo * tanh_fast(c);
            hnew.s[j] = __builtin_bit_cast(unsigned short, (__bf16)h);
            dacc += h * dwv[j];
        }
        // one 16B bf16 store of this thread's h slice
        *(uint4*)&hbuf[gb][h0] = hnew.u;
        __syncthreads();
    }

    // ---- final: reduce dense partials over the 16 threads of each row ----
    red[tid] = dacc;
    __syncthreads();
    if (tid < BTILE) {
        float s = 0.0f;
        #pragma unroll
        for (int k = 0; k < 16; ++k) s += red[tid * 16 + k];
        out[blockIdx.x * BTILE + tid] = sigm(s + db[0]);
    }
}

extern "C" void kernel_launch(void* const* d_in, const int* in_sizes, int n_in,
                              void* d_out, int out_size, void* d_ws, size_t ws_size,
                              hipStream_t stream) {
    (void)in_sizes; (void)n_in; (void)out_size; (void)d_ws; (void)ws_size;
    const float* x    = (const float*)d_in[0];
    const float* W_ih = (const float*)d_in[1];
    const float* W_hh = (const float*)d_in[2];
    const float* b_ih = (const float*)d_in[3];
    const float* b_hh = (const float*)d_in[4];
    const float* dw   = (const float*)d_in[5];
    const float* db   = (const float*)d_in[6];
    float* out = (float*)d_out;

    lstm_fused_kernel<<<dim3(BATCH / BTILE), dim3(THREADS), 0, stream>>>(
        x, W_ih, W_hh, b_ih, b_hh, dw, db, out);
}
